// SingleBlock_78812649882395
// MI455X (gfx1250) — matrix operand order repaired and checked by reading, not compiled, not run on hardware
//
#include <hip/hip_runtime.h>
#include <math.h>

// ---------------- problem constants ----------------
#define B_   128
#define NO_  100
#define L_   36
#define VS_  2048
#define TS_  1024
#define D_   1024
#define H_   8
#define DH_  128
#define TD_  3072              // 3*D
#define SCALE_ 0.088388347648318447f   // 1/sqrt(128)

typedef __bf16 bf16_t;
typedef __attribute__((ext_vector_type(16))) __bf16 v16bf;
typedef __attribute__((ext_vector_type(8)))  float  v8f;

__device__ __forceinline__ bf16_t f2bf(float f) {
  unsigned u = __builtin_bit_cast(unsigned, f);
  u += 0x7FFFu + ((u >> 16) & 1u);           // round-to-nearest-even
  return __builtin_bit_cast(bf16_t, (unsigned short)(u >> 16));
}

// ---------------- WMMA bf16 GEMM: C[M,N] = (relu?)A[M,K] * B[K,N] (+bias) (+=C) ----
// 128x128 block tile, BK=32, 8 waves (wave32) in a 4x2 grid; each wave owns a
// 32x64 sub-tile = 2x4 WMMA tiles -> 8 v_wmma per wave per k-step.
#define BM 128
#define BN 128
#define BK 32
#define LDSS 34   // padded LDS row stride (bf16 elems); 17 banks, coprime w/ 64

__global__ __launch_bounds__(256)
void gemm_bf16_wmma(const float* __restrict__ A, const float* __restrict__ Bw,
                    const float* __restrict__ bias, float* __restrict__ C,
                    int M, int N, int K, int relu_a, int accumulate, int has_bias)
{
  __shared__ bf16_t As[BM * LDSS];
  __shared__ bf16_t Bs[BN * LDSS];   // B tile stored transposed: Bs[n][k]

  const int tid  = threadIdx.x;
  const int wave = tid >> 5;
  const int lane = tid & 31;
  const int g    = lane >> 4;        // lane group (0: lanes 0-15, 1: lanes 16-31)
  const int lr   = lane & 15;
  const int wm   = (wave >> 1) * 32; // wave's 32x64 sub-tile origin
  const int wn   = (wave & 1) * 64;
  const int bm   = blockIdx.x * BM;
  const int bn   = blockIdx.y * BN;

  v8f acc[2][4];
#pragma unroll
  for (int i = 0; i < 2; ++i)
#pragma unroll
    for (int j = 0; j < 4; ++j)
      acc[i][j] = (v8f){0.f,0.f,0.f,0.f,0.f,0.f,0.f,0.f};

  for (int k0 = 0; k0 < K; k0 += BK) {
    // ---- stage A tile (128x32 f32 -> bf16), 4 float4 per thread
#pragma unroll
    for (int i = 0; i < 4; ++i) {
      int idx = tid + i * 256;
      int r = idx >> 3;
      int c = (idx & 7) << 2;
      const float* ap = A + (size_t)(bm + r) * K + (k0 + c);
      float4 x = *(const float4*)ap;
      if (k0 + BK < K) __builtin_prefetch(ap + BK, 0, 1);   // global_prefetch_b8
      if (relu_a) {
        x.x = fmaxf(x.x, 0.f); x.y = fmaxf(x.y, 0.f);
        x.z = fmaxf(x.z, 0.f); x.w = fmaxf(x.w, 0.f);
      }
      bf16_t* d = &As[r * LDSS + c];
      d[0] = f2bf(x.x); d[1] = f2bf(x.y); d[2] = f2bf(x.z); d[3] = f2bf(x.w);
    }
    // ---- stage B tile (32x128 f32 -> bf16, transposed into Bs[n][k])
#pragma unroll
    for (int i = 0; i < 4; ++i) {
      int idx = tid + i * 256;
      int r = idx >> 5;            // k within tile
      int c = (idx & 31) << 2;     // n within tile
      const float* bp = Bw + (size_t)(k0 + r) * N + (bn + c);
      float4 x = *(const float4*)bp;
      if (k0 + BK < K) __builtin_prefetch(bp + (size_t)BK * N, 0, 1);
      Bs[(c + 0) * LDSS + r] = f2bf(x.x);
      Bs[(c + 1) * LDSS + r] = f2bf(x.y);
      Bs[(c + 2) * LDSS + r] = f2bf(x.z);
      Bs[(c + 3) * LDSS + r] = f2bf(x.w);
    }
    __syncthreads();

    // ---- build fragments per documented CDNA5 16-bit layouts
    v16bf afrag[2], bfrag[4];
#pragma unroll
    for (int ti = 0; ti < 2; ++ti) {
      const bf16_t* ap = &As[(wm + ti * 16 + lr) * LDSS];
#pragma unroll
      for (int j = 0; j < 16; ++j) {
        // A 16x32: elem j -> K = j+8g (j<8) else j+8+8g
        int kk = (j < 8) ? (j + 8 * g) : (j + 8 + 8 * g);
        afrag[ti][j] = ap[kk];
      }
    }
#pragma unroll
    for (int tj = 0; tj < 4; ++tj) {
      const bf16_t* bp = &Bs[(wn + tj * 16 + lr) * LDSS];
#pragma unroll
      for (int j = 0; j < 16; ++j)          // B 32x16: elem j -> K = j+8g (j<8) else j+8+8g
        bfrag[tj][j] = bp[(j < 8) ? (j + 8 * g) : (j + 8 + 8 * g)];
    }
#pragma unroll
    for (int ti = 0; ti < 2; ++ti)
#pragma unroll
      for (int tj = 0; tj < 4; ++tj)
        acc[ti][tj] = __builtin_amdgcn_wmma_f32_16x16x32_bf16(
            false, afrag[ti], false, bfrag[tj], (short)0, acc[ti][tj], false, false);
    __syncthreads();
  }

  // ---- epilogue: C layout elem r -> M = r + 8g, N = lr
#pragma unroll
  for (int ti = 0; ti < 2; ++ti) {
#pragma unroll
    for (int tj = 0; tj < 4; ++tj) {
      int col = bn + wn + tj * 16 + lr;
      float bv = has_bias ? bias[col] : 0.f;
#pragma unroll
      for (int r = 0; r < 8; ++r) {
        int row = bm + wm + ti * 16 + 8 * g + r;
        size_t o = (size_t)row * N + col;
        float val = acc[ti][tj][r] + bv;
        if (accumulate) val += C[o];
        C[o] = val;
      }
    }
  }
}

// ---------------- fused masked-softmax attention (fp32, wave32) ----------------
// One wave per query row. Q rows in Q[b,Nq,3D] at col qoff+h*DH; K/V rows in
// KV[b,Nk,3D] at koff/voff. Optional gate[B,D]: q *= g^2 (== gating q and k).
__global__ __launch_bounds__(128)
void attn_kernel(const float* __restrict__ Q, const float* __restrict__ KV,
                 const float* __restrict__ key_mask, const float* __restrict__ gate,
                 float* __restrict__ Out, int Nq, int Nk, int qoff, int koff, int voff)
{
  const int lane = threadIdx.x & 31;
  const int wv   = threadIdx.x >> 5;
  const int n    = blockIdx.x * 4 + wv;
  const int h    = blockIdx.y;
  const int b    = blockIdx.z;
  if (n >= Nq) return;

  const float NEG_INF = -__builtin_inff();

  const float* qp = Q + ((size_t)b * Nq + n) * TD_ + qoff + h * DH_;
  float qv[4];
#pragma unroll
  for (int r = 0; r < 4; ++r) qv[r] = qp[lane + 32 * r];
  if (gate) {
    const float* gp = gate + (size_t)b * D_ + h * DH_;
#pragma unroll
    for (int r = 0; r < 4; ++r) { float gg = gp[lane + 32 * r]; qv[r] *= gg * gg; }
  }

  float sc[4];
  float mx = NEG_INF;
#pragma unroll
  for (int j = 0; j < 4; ++j) {
    float mine = NEG_INF;
    if (j * 32 < Nk) {
      for (int mm = 0; mm < 32; ++mm) {
        int m = j * 32 + mm;
        if (m >= Nk) break;
        const float* kp = KV + ((size_t)b * Nk + m) * TD_ + koff + h * DH_;
        float s = qv[0] * kp[lane] + qv[1] * kp[lane + 32]
                + qv[2] * kp[lane + 64] + qv[3] * kp[lane + 96];
#pragma unroll
        for (int off = 16; off > 0; off >>= 1) s += __shfl_xor(s, off, 32);
        s *= SCALE_;
        if (!(key_mask[(size_t)b * Nk + m] > 0.f)) s = NEG_INF;
        if (mm == lane) mine = s;
        mx = fmaxf(mx, s);
      }
    }
    sc[j] = mine;
  }

  float e[4];
  float denom = 0.f;
#pragma unroll
  for (int j = 0; j < 4; ++j) { e[j] = __expf(sc[j] - mx); denom += e[j]; }
#pragma unroll
  for (int off = 16; off > 0; off >>= 1) denom += __shfl_xor(denom, off, 32);
  float inv = 1.f / denom;

  float av[4] = {0.f, 0.f, 0.f, 0.f};
#pragma unroll
  for (int j = 0; j < 4; ++j) {
    if (j * 32 < Nk) {
      float ej = e[j];
      for (int mm = 0; mm < 32; ++mm) {
        int m = j * 32 + mm;
        if (m >= Nk) break;
        float p = __shfl(ej, mm, 32) * inv;
        const float* vp = KV + ((size_t)b * Nk + m) * TD_ + voff + h * DH_;
        av[0] += p * vp[lane];
        av[1] += p * vp[lane + 32];
        av[2] += p * vp[lane + 64];
        av[3] += p * vp[lane + 96];
      }
    }
  }
  float* op = Out + ((size_t)b * Nq + n) * D_ + h * DH_;
#pragma unroll
  for (int r = 0; r < 4; ++r) op[lane + 32 * r] = av[r];
}

// ---------------- small elementwise kernels ----------------
__global__ void rowscale_kernel(float* __restrict__ x, const float* __restrict__ m,
                                long long rows, int cols) {
  long long i = (long long)blockIdx.x * blockDim.x + threadIdx.x;
  if (i < rows * (long long)cols) x[i] *= m[i / cols];
}

__global__ void gatefin_kernel(float* __restrict__ x, int n) {
  int i = blockIdx.x * blockDim.x + threadIdx.x;
  if (i < n) x[i] = 1.f + 1.f / (1.f + __expf(-x[i]));
}

__global__ void add_inplace_kernel(float* __restrict__ d, const float* __restrict__ a,
                                   long long n) {
  long long i = (long long)blockIdx.x * blockDim.x + threadIdx.x;
  if (i < n) d[i] += a[i];
}

__global__ void masked_mean_kernel(const float* __restrict__ x, const float* __restrict__ m,
                                   float* __restrict__ out, int nrows) {
  int i = blockIdx.x * blockDim.x + threadIdx.x;   // over B_*D_
  if (i >= B_ * D_) return;
  int b = i / D_, d = i % D_;
  float s = 0.f, ms = 0.f;
  for (int r = 0; r < nrows; ++r) {
    float mv = m[b * nrows + r];
    s  += x[((size_t)b * nrows + r) * D_ + d] * mv;
    ms += mv;
  }
  out[i] = s / ms;
}

// ---------------- host orchestration ----------------
extern "C" void kernel_launch(void* const* d_in, const int* in_sizes, int n_in,
                              void* d_out, int out_size, void* d_ws, size_t ws_size,
                              hipStream_t stream) {
  (void)in_sizes; (void)n_in; (void)out_size; (void)ws_size;

  const float* v_in   = (const float*)d_in[0];
  const float* t_in   = (const float*)d_in[1];
  const float* vm     = (const float*)d_in[2];
  const float* tmk    = (const float*)d_in[3];
  const float* vlin_w = (const float*)d_in[4];
  const float* vlin_b = (const float*)d_in[5];
  const float* tlin_w = (const float*)d_in[6];
  const float* tlin_b = (const float*)d_in[7];
  const float* iv_w   = (const float*)d_in[8];
  const float* iv_b   = (const float*)d_in[9];
  const float* it_w   = (const float*)d_in[10];
  const float* it_b   = (const float*)d_in[11];
  const float* ivo_w  = (const float*)d_in[12];
  const float* ivo_b  = (const float*)d_in[13];
  const float* ito_w  = (const float*)d_in[14];
  const float* ito_b  = (const float*)d_in[15];
  const float* gv_w   = (const float*)d_in[16];
  const float* gv_b   = (const float*)d_in[17];
  const float* gt_w   = (const float*)d_in[18];
  const float* gt_b   = (const float*)d_in[19];
  const float* av_w   = (const float*)d_in[20];
  const float* av_b   = (const float*)d_in[21];
  const float* at_w   = (const float*)d_in[22];
  const float* at_b   = (const float*)d_in[23];
  const float* avo_w  = (const float*)d_in[24];
  const float* avo_b  = (const float*)d_in[25];
  const float* ato_w  = (const float*)d_in[26];
  const float* ato_b  = (const float*)d_in[27];

  const int MV = B_ * NO_;   // 12800 rows (divisible by 128)
  const int MT = B_ * L_;    // 4608 rows  (divisible by 128)

  float* ws = (float*)d_ws;
  size_t off = 0;
  float* vA    = ws + off; off += (size_t)MV * D_;
  float* vB    = ws + off; off += (size_t)MV * D_;
  float* tA    = ws + off; off += (size_t)MT * D_;
  float* tB    = ws + off; off += (size_t)MT * D_;
  float* vqkv  = ws + off; off += (size_t)MV * TD_;
  float* tqkv  = ws + off; off += (size_t)MT * TD_;
  float* vup   = ws + off; off += (size_t)MV * D_;
  float* tup   = ws + off; off += (size_t)MT * D_;
  float* vmean = ws + off; off += (size_t)B_ * D_;
  float* tmean = ws + off; off += (size_t)B_ * D_;
  float* gvb   = ws + off; off += (size_t)B_ * D_;   // v4t gate (gates text)
  float* gtb   = ws + off; off += (size_t)B_ * D_;   // t4v gate (gates vision)

  auto gemm = [&](const float* A, const float* Bw, const float* bias, float* C,
                  int M, int N, int K, bool relu, bool accum) {
    dim3 grid((unsigned)(M / BM), (unsigned)(N / BN));
    gemm_bf16_wmma<<<grid, dim3(256), 0, stream>>>(A, Bw, bias, C, M, N, K,
                                                   relu ? 1 : 0, accum ? 1 : 0,
                                                   bias ? 1 : 0);
  };
  auto rowscale = [&](float* x, const float* m, long long rows, int cols) {
    long long total = rows * (long long)cols;
    rowscale_kernel<<<dim3((unsigned)((total + 255) / 256)), dim3(256), 0, stream>>>(
        x, m, rows, cols);
  };
  auto attn = [&](const float* Qb, const float* KVb, const float* km, const float* gate,
                  float* out, int Nq, int Nk) {
    attn_kernel<<<dim3((unsigned)(Nq / 4), H_, B_), dim3(128), 0, stream>>>(
        Qb, KVb, km, gate, out, Nq, Nk, D_, 0, 2 * D_);
  };

  // ---- initial projections
  gemm(v_in, vlin_w, vlin_b, vA, MV, D_, VS_, false, false);
  gemm(t_in, tlin_w, tlin_b, tA, MT, D_, TS_, false, false);

  float* vc = vA; float* vn = vB;
  float* tc = tA; float* tn = tB;

  for (int blk = 0; blk < 2; ++blk) {
    // ======== InterModalityUpdate ========
    rowscale(vc, vm, MV, D_);
    rowscale(tc, tmk, MT, D_);
    gemm(vc, iv_w, iv_b, vqkv, MV, TD_, D_, true, false);
    gemm(tc, it_w, it_b, tqkv, MT, TD_, D_, true, false);
    attn(vqkv, tqkv, tmk, nullptr, vup, NO_, L_);   // t2v -> v_up
    attn(tqkv, vqkv, vm,  nullptr, tup, L_, NO_);   // v2t -> t_up
    // concat([v, v_up]) @ ivo_w == v @ W_top + v_up @ W_bot
    gemm(vc,  ivo_w,                   ivo_b,  vn, MV, D_, D_, false, false);
    gemm(vup, ivo_w + (size_t)D_ * D_, nullptr, vn, MV, D_, D_, false, true);
    gemm(tc,  ito_w,                   ito_b,  tn, MT, D_, D_, false, false);
    gemm(tup, ito_w + (size_t)D_ * D_, nullptr, tn, MT, D_, D_, false, true);
    { float* s = vc; vc = vn; vn = s; }
    { float* s = tc; tc = tn; tn = s; }

    // ======== DyIntraModalityUpdate ========
    masked_mean_kernel<<<dim3((B_ * D_ + 255) / 256), dim3(256), 0, stream>>>(vc, vm, vmean, NO_);
    masked_mean_kernel<<<dim3((B_ * D_ + 255) / 256), dim3(256), 0, stream>>>(tc, tmk, tmean, L_);
    gemm(vmean, gv_w, gv_b, gvb, B_, D_, D_, true, false);
    gemm(tmean, gt_w, gt_b, gtb, B_, D_, D_, true, false);
    gatefin_kernel<<<dim3((B_ * D_ + 255) / 256), dim3(256), 0, stream>>>(gvb, B_ * D_);
    gatefin_kernel<<<dim3((B_ * D_ + 255) / 256), dim3(256), 0, stream>>>(gtb, B_ * D_);
    gemm(vc, av_w, av_b, vqkv, MV, TD_, D_, true, false);
    rowscale(vqkv, vm, MV, TD_);
    gemm(tc, at_w, at_b, tqkv, MT, TD_, D_, true, false);
    rowscale(tqkv, tmk, MT, TD_);
    attn(vqkv, vqkv, vm,  gtb, vup, NO_, NO_);      // gate_v = 1 + t4v_gate
    attn(tqkv, tqkv, tmk, gvb, tup, L_, L_);        // gate_t = 1 + v4t_gate
    add_inplace_kernel<<<dim3((unsigned)(((long long)MV * D_ + 255) / 256)), dim3(256), 0, stream>>>(
        vup, vc, (long long)MV * D_);
    add_inplace_kernel<<<dim3((unsigned)(((long long)MT * D_ + 255) / 256)), dim3(256), 0, stream>>>(
        tup, tc, (long long)MT * D_);
    gemm(vup, avo_w, avo_b, vn, MV, D_, D_, false, false);
    gemm(tup, ato_w, ato_b, tn, MT, D_, D_, false, false);
    { float* s = vc; vc = vn; vn = s; }
    { float* s = tc; tc = tn; tn = s; }
  }

  // ---- outputs: (v, t) concatenated flat
  hipMemcpyAsync(d_out, vc, (size_t)MV * D_ * sizeof(float),
                 hipMemcpyDeviceToDevice, stream);
  hipMemcpyAsync((float*)d_out + (size_t)MV * D_, tc, (size_t)MT * D_ * sizeof(float),
                 hipMemcpyDeviceToDevice, stream);
}
